// GAT_76261439308198
// MI455X (gfx1250) — matrix operand order repaired and checked by reading, not compiled, not run on hardware
//
#include <hip/hip_runtime.h>
#include <hip/hip_bf16.h>

#define N_NODES 50000
#define N_EDGES 800000
#define ED_F    32
#define LDST    132   // padded LDS row stride in floats (kills 16-way bank conflicts)

typedef __attribute__((ext_vector_type(16))) __bf16 v16bf;
typedef __attribute__((ext_vector_type(8)))  __bf16 v8bf;
typedef __attribute__((ext_vector_type(8)))  float  v8f;

// ---- order-preserving float<->uint encode for atomic max (enc(-inf)=0x007FFFFF > 0 => memset-0 init ok)
static __device__ __forceinline__ unsigned encf(float x) {
  unsigned b = __float_as_uint(x);
  return (b & 0x80000000u) ? ~b : (b | 0x80000000u);
}
static __device__ __forceinline__ float decf(unsigned k) {
  return __uint_as_float((k & 0x80000000u) ? (k & 0x7FFFFFFFu) : ~k);
}

// ---- A-fragment loader: contiguous-k A 16x32 order (same as packed B) (lanes 0-15: K kb..kb+15 with kb=0; lanes 16-31: kb=16)
static __device__ __forceinline__ v16bf load_afrag(const __bf16* rowp, int lane) {
  const int kb = (lane < 16) ? 0 : 16;
  const v8bf lo = *(const v8bf*)(rowp + kb);
  const v8bf hi = *(const v8bf*)(rowp + kb + 8);
  v16bf a;
#pragma unroll
  for (int i = 0; i < 8; ++i) { a[i] = lo[i]; a[i + 8] = hi[i]; }
  return a;
}

// =====================  prep kernels  =====================
__global__ void cvt_bf16(const float* __restrict__ src, __bf16* __restrict__ dst, int n) {
  int i  = blockIdx.x * blockDim.x + threadIdx.x;
  int st = blockDim.x * gridDim.x;
  for (; i < n; i += st) dst[i] = (__bf16)src[i];
}

// Pre-swizzle weights into per-lane WMMA B-fragment order:
//   B 32x16 bf16 fragment: lane L holds col = 16*ct + (L&15), K = (L<16?0:16)+i, i=0..15 (contiguous 32B/lane)
__global__ void pack_weights_l1(const float* __restrict__ Wq, const float* __restrict__ Wk,
                                const float* __restrict__ Wv, const float* __restrict__ Ws,
                                const float* __restrict__ bq, const float* __restrict__ bk,
                                const float* __restrict__ bv, const float* __restrict__ bs,
                                const float* __restrict__ We,
                                __bf16* __restrict__ Wfrag, float* __restrict__ biaspack,
                                __bf16* __restrict__ Wefrag) {
  const int tid = blockIdx.x * blockDim.x + threadIdx.x;
  const int nt  = blockDim.x * gridDim.x;
  // Wfrag: [ks(4)][ct(32)][lane(32)][i(16)] ; ct 0..7=Wq, 8..15=Wk, 16..23=Wv, 24..31=Ws
  for (int idx = tid; idx < 4 * 32 * 32 * 16; idx += nt) {
    int i = idx & 15, lane = (idx >> 4) & 31, ct = (idx >> 9) & 31, ks = idx >> 14;
    int m = ct >> 3;
    int colm = 16 * (ct & 7) + (lane & 15);
    int krow = 32 * ks + ((lane < 16) ? 0 : 16) + i;
    const float* W = (m == 0) ? Wq : (m == 1) ? Wk : (m == 2) ? Wv : Ws;
    Wfrag[idx] = (__bf16)W[krow * 128 + colm];
  }
  for (int c = tid; c < 512; c += nt) {
    int m = c >> 7;
    const float* b = (m == 0) ? bq : (m == 1) ? bk : (m == 2) ? bv : bs;
    biaspack[c] = b[c & 127];
  }
  // Wefrag: [ct(8)][lane(32)][i(16)], K=32 single step, We is [32][128]
  for (int idx = tid; idx < 8 * 32 * 16; idx += nt) {
    int i = idx & 15, lane = (idx >> 4) & 31, ct = idx >> 9;
    int col  = 16 * ct + (lane & 15);
    int krow = ((lane < 16) ? 0 : 16) + i;
    Wefrag[idx] = (__bf16)We[krow * 128 + col];
  }
}

// =====================  layer-1 node GEMM: [N,128]bf16 x [128,512]bf16 -> q|k|v|s f32  =====================
__global__ __launch_bounds__(256) void node_gemm_l1(const __bf16* __restrict__ xb,
                                                    const __bf16* __restrict__ Wfrag,
                                                    const float* __restrict__ biaspack,
                                                    float* __restrict__ nodebuf) {
  const int lane = threadIdx.x & 31;
  const int w    = threadIdx.x >> 5;       // 8 waves: wave w owns col-tiles 4w..4w+3
  const int rt   = blockIdx.x;             // 3125 row-tiles of 16 (exact)
  const int row  = rt * 16 + (lane & 15);
  v8f acc[4];
#pragma unroll
  for (int j = 0; j < 4; ++j)
#pragma unroll
    for (int r = 0; r < 8; ++r) acc[j][r] = 0.f;
#pragma unroll
  for (int ks = 0; ks < 4; ++ks) {
    v16bf A = load_afrag(xb + (size_t)row * 128 + ks * 32, lane);
#pragma unroll
    for (int j = 0; j < 4; ++j) {
      const int ct = 4 * w + j;
      v16bf B = *(const v16bf*)(Wfrag + ((size_t)(ks * 32 + ct) * 32 + lane) * 16);
      acc[j] = __builtin_amdgcn_wmma_f32_16x16x32_bf16(false, A, false, B, (short)0, acc[j], false, false);
    }
  }
  const int rbase = rt * 16 + ((lane < 16) ? 0 : 8);
#pragma unroll
  for (int j = 0; j < 4; ++j) {
    const int col = 16 * (4 * w + j) + (lane & 15);
    const float bias = biaspack[col];
#pragma unroll
    for (int r = 0; r < 8; ++r)
      nodebuf[(size_t)(rbase + r) * 512 + col] = acc[j][r] + bias;
  }
}

// =====================  fused edge-embedding tile (WMMA -> LDS)  =====================
static __device__ __forceinline__ void compute_etile(const __bf16* __restrict__ eab, int tile,
                                                     const v16bf* Bf, const float* biasv,
                                                     float* lds, int lane) {
  const int erow = tile * 16 + (lane & 15);
  v16bf A = load_afrag(eab + (size_t)erow * ED_F, lane);
  const int rowoff = (lane < 16) ? 0 : 8;
  const int colb   = lane & 15;
#pragma unroll
  for (int ct = 0; ct < 8; ++ct) {
    v8f c;
#pragma unroll
    for (int r = 0; r < 8; ++r) c[r] = biasv[ct];
    c = __builtin_amdgcn_wmma_f32_16x16x32_bf16(false, A, false, Bf[ct], (short)0, c, false, false);
    const int col = 16 * ct + colb;
#pragma unroll
    for (int r = 0; r < 8; ++r) lds[(rowoff + r) * LDST + col] = c[r];
  }
  __builtin_amdgcn_wave_barrier();
  asm volatile("s_wait_dscnt 0x0" ::: "memory");   // per-wave LDS RAW fence (CDNA5 split counters)
  __builtin_amdgcn_wave_barrier();
}

// =====================  layer-1 edge pass 1: alpha + segment max  =====================
__global__ __launch_bounds__(128) void edge_attn_l1(const __bf16* __restrict__ eab,
                                                    const __bf16* __restrict__ Wefrag,
                                                    const float* __restrict__ be1,
                                                    const int* __restrict__ src, const int* __restrict__ dst,
                                                    const float* __restrict__ nodebuf,
                                                    float* __restrict__ alpha1, unsigned int* __restrict__ amax1) {
  __shared__ float etile[4][16 * LDST];
  const int lane = threadIdx.x & 31;
  const int w    = threadIdx.x >> 5;
  float* lds = etile[w];
  v16bf Bf[8];
  float biasv[8];
#pragma unroll
  for (int ct = 0; ct < 8; ++ct) {
    Bf[ct]    = *(const v16bf*)(Wefrag + (size_t)(ct * 32 + lane) * 16);
    biasv[ct] = be1[16 * ct + (lane & 15)];
  }
  const int nT = N_EDGES / 16;
  const int nW = gridDim.x * 4;
  const float scale = 0.17677669529663687f;  // 1/sqrt(32)
  for (int tile = blockIdx.x * 4 + w; tile < nT; tile += nW) {
    if (tile + nW < nT) __builtin_prefetch(eab + (size_t)(tile + nW) * 16 * ED_F, 0, 1);
    compute_etile(eab, tile, Bf, biasv, lds, lane);
    const int m = lane & 15, half = lane >> 4;
    const int e = tile * 16 + m;
    const int s = src[e], d = dst[e];
    const float4* qp = (const float4*)(nodebuf + (size_t)d * 512 + 64 * half);
    const float4* kp = (const float4*)(nodebuf + (size_t)s * 512 + 128 + 64 * half);
    const float4* ep = (const float4*)(lds + m * LDST + 64 * half);
    float a0 = 0.f, a1 = 0.f;
#pragma unroll
    for (int j = 0; j < 8; ++j) {
      float4 q4 = qp[j], k4 = kp[j], e4 = ep[j];
      float t = q4.x * (k4.x + e4.x) + q4.y * (k4.y + e4.y) +
                q4.z * (k4.z + e4.z) + q4.w * (k4.w + e4.w);
      if (j < 4) a0 += t; else a1 += t;
    }
    a0 *= scale; a1 *= scale;
    const int h0 = 2 * half;
    alpha1[(size_t)e * 4 + h0]     = a0;
    alpha1[(size_t)e * 4 + h0 + 1] = a1;
    atomicMax(amax1 + (size_t)d * 4 + h0,     encf(a0));
    atomicMax(amax1 + (size_t)d * 4 + h0 + 1, encf(a1));
  }
}

// =====================  layer-1 edge pass 2: exp + scatter-accumulate  =====================
__global__ __launch_bounds__(128) void edge_agg_l1(const __bf16* __restrict__ eab,
                                                   const __bf16* __restrict__ Wefrag,
                                                   const float* __restrict__ be1,
                                                   const int* __restrict__ src, const int* __restrict__ dst,
                                                   const float* __restrict__ nodebuf,
                                                   const float* __restrict__ alpha1,
                                                   const unsigned int* __restrict__ amax1,
                                                   float* __restrict__ denom1, float* __restrict__ acc1) {
  __shared__ float etile[4][16 * LDST];
  const int lane = threadIdx.x & 31;
  const int w    = threadIdx.x >> 5;
  float* lds = etile[w];
  v16bf Bf[8];
  float biasv[8];
#pragma unroll
  for (int ct = 0; ct < 8; ++ct) {
    Bf[ct]    = *(const v16bf*)(Wefrag + (size_t)(ct * 32 + lane) * 16);
    biasv[ct] = be1[16 * ct + (lane & 15)];
  }
  const int nT = N_EDGES / 16;
  const int nW = gridDim.x * 4;
  for (int tile = blockIdx.x * 4 + w; tile < nT; tile += nW) {
    if (tile + nW < nT) __builtin_prefetch(eab + (size_t)(tile + nW) * 16 * ED_F, 0, 1);
    compute_etile(eab, tile, Bf, biasv, lds, lane);
    const int m = lane & 15, half = lane >> 4;
    const int e = tile * 16 + m;
    const int s = src[e], d = dst[e];
    const int h0 = 2 * half;
    const float am0 = decf(amax1[(size_t)d * 4 + h0]);
    const float am1 = decf(amax1[(size_t)d * 4 + h0 + 1]);
    const float ex0 = __expf(alpha1[(size_t)e * 4 + h0]     - am0);
    const float ex1 = __expf(alpha1[(size_t)e * 4 + h0 + 1] - am1);
    atomicAdd(denom1 + (size_t)d * 4 + h0,     ex0);
    atomicAdd(denom1 + (size_t)d * 4 + h0 + 1, ex1);
    const float* vp = nodebuf + (size_t)s * 512 + 256 + 64 * half;
    const float* ep = lds + m * LDST + 64 * half;
    float* ap = acc1 + (size_t)d * 128 + 64 * half;
#pragma unroll 8
    for (int j = 0; j < 32; ++j) atomicAdd(ap + j,      ex0 * (vp[j]      + ep[j]));
#pragma unroll 8
    for (int j = 0; j < 32; ++j) atomicAdd(ap + 32 + j, ex1 * (vp[32 + j] + ep[32 + j]));
  }
}

// =====================  layer-1 finalize: divide + skip + LayerNorm + ELU  =====================
__global__ __launch_bounds__(256) void node_finish_l1(const float* __restrict__ acc1,
                                                      const float* __restrict__ denom1,
                                                      const float* __restrict__ nodebuf,
                                                      const float* __restrict__ ln_g,
                                                      const float* __restrict__ ln_b,
                                                      float* __restrict__ hout) {
  const int lane = threadIdx.x & 31;
  const int n = (blockIdx.x * blockDim.x + threadIdx.x) >> 5;  // wave per node
  if (n >= N_NODES) return;
  float v[4];
#pragma unroll
  for (int j = 0; j < 4; ++j) {
    const int c = 32 * j + lane;   // head = c/32 = j
    const float dn = denom1[(size_t)n * 4 + j] + 1e-16f;
    v[j] = acc1[(size_t)n * 128 + c] / dn + nodebuf[(size_t)n * 512 + 384 + c];
  }
  float s = v[0] + v[1] + v[2] + v[3];
#pragma unroll
  for (int k = 16; k; k >>= 1) s += __shfl_xor(s, k, 32);
  const float mu = s * (1.0f / 128.0f);
  float vs = 0.f;
#pragma unroll
  for (int j = 0; j < 4; ++j) { const float dlt = v[j] - mu; vs += dlt * dlt; }
#pragma unroll
  for (int k = 16; k; k >>= 1) vs += __shfl_xor(vs, k, 32);
  const float rstd = rsqrtf(vs * (1.0f / 128.0f) + 1e-5f);
#pragma unroll
  for (int j = 0; j < 4; ++j) {
    const int c = 32 * j + lane;
    float y = (v[j] - mu) * rstd * ln_g[c] + ln_b[c];
    y = (y > 0.f) ? y : (__expf(y) - 1.f);  // ELU
    hout[(size_t)n * 128 + c] = y;
  }
}

// =====================  layer-2 node linear (width 1 -> scalar dots)  =====================
__global__ __launch_bounds__(256) void node_lin_l2(const float* __restrict__ h,
                                                   const float* __restrict__ Wq, const float* __restrict__ bq,
                                                   const float* __restrict__ Wk, const float* __restrict__ bk,
                                                   const float* __restrict__ Wv, const float* __restrict__ bv,
                                                   const float* __restrict__ Ws, const float* __restrict__ bs,
                                                   float* __restrict__ q2, float* __restrict__ k2,
                                                   float* __restrict__ v2, float* __restrict__ s2) {
  const int lane = threadIdx.x & 31;
  const int n = (blockIdx.x * blockDim.x + threadIdx.x) >> 5;
  if (n >= N_NODES) return;
  float pq = 0.f, pk = 0.f, pv = 0.f, ps = 0.f;
#pragma unroll
  for (int j = 0; j < 4; ++j) {
    const int c = 32 * j + lane;
    const float xv = h[(size_t)n * 128 + c];
    pq += xv * Wq[c]; pk += xv * Wk[c]; pv += xv * Wv[c]; ps += xv * Ws[c];
  }
#pragma unroll
  for (int k = 16; k; k >>= 1) {
    pq += __shfl_xor(pq, k, 32); pk += __shfl_xor(pk, k, 32);
    pv += __shfl_xor(pv, k, 32); ps += __shfl_xor(ps, k, 32);
  }
  if (lane == 0) {
    q2[n] = pq + bq[0]; k2[n] = pk + bk[0];
    v2[n] = pv + bv[0]; s2[n] = ps + bs[0];
  }
}

// =====================  layer-2 edge passes  =====================
__global__ __launch_bounds__(256) void edge_l2_p1(const float* __restrict__ ea,
                                                  const float* __restrict__ We2, const float* __restrict__ be2,
                                                  const int* __restrict__ src, const int* __restrict__ dst,
                                                  const float* __restrict__ q2, const float* __restrict__ k2,
                                                  float* __restrict__ e2buf, float* __restrict__ alpha2,
                                                  unsigned int* __restrict__ amax2) {
  const int e = blockIdx.x * blockDim.x + threadIdx.x;
  if (e >= N_EDGES) return;
  const float4* ap = (const float4*)(ea + (size_t)e * 32);
  const float4* wp = (const float4*)We2;
  float acc = 0.f;
#pragma unroll
  for (int j = 0; j < 8; ++j) {
    float4 a = ap[j], w = wp[j];
    acc += a.x * w.x + a.y * w.y + a.z * w.z + a.w * w.w;
  }
  const float ev = acc + be2[0];
  const int s = src[e], d = dst[e];
  const float al = q2[d] * (k2[s] + ev);   // /sqrt(OUT)=1
  e2buf[e] = ev; alpha2[e] = al;
  atomicMax(amax2 + d, encf(al));
}

__global__ __launch_bounds__(256) void edge_l2_p2(const int* __restrict__ src, const int* __restrict__ dst,
                                                  const float* __restrict__ v2,
                                                  const float* __restrict__ e2buf,
                                                  const float* __restrict__ alpha2,
                                                  const unsigned int* __restrict__ amax2,
                                                  float* __restrict__ denom2, float* __restrict__ acc2) {
  const int e = blockIdx.x * blockDim.x + threadIdx.x;
  if (e >= N_EDGES) return;
  const int s = src[e], d = dst[e];
  const float ex = __expf(alpha2[e] - decf(amax2[d]));
  atomicAdd(denom2 + d, ex);
  atomicAdd(acc2 + d, ex * (v2[s] + e2buf[e]));
}

__global__ __launch_bounds__(256) void node_final(const float* __restrict__ acc2,
                                                  const float* __restrict__ denom2,
                                                  const float* __restrict__ s2,
                                                  float* __restrict__ out) {
  const int n = blockIdx.x * blockDim.x + threadIdx.x;
  if (n >= N_NODES) return;
  const float logit = acc2[n] / (denom2[n] + 1e-16f) + s2[n];
  out[n] = 1.0f / (1.0f + __expf(-logit));
}

// =====================  launcher  =====================
static inline size_t alignup(size_t x) { return (x + 255) & ~(size_t)255; }

extern "C" void kernel_launch(void* const* d_in, const int* in_sizes, int n_in,
                              void* d_out, int out_size, void* d_ws, size_t ws_size,
                              hipStream_t stream) {
  const float* x    = (const float*)d_in[0];
  const int*   eidx = (const int*)d_in[1];
  const float* ea   = (const float*)d_in[2];
  const float* Wq1 = (const float*)d_in[3];  const float* bq1 = (const float*)d_in[4];
  const float* Wk1 = (const float*)d_in[5];  const float* bk1 = (const float*)d_in[6];
  const float* Wv1 = (const float*)d_in[7];  const float* bv1 = (const float*)d_in[8];
  const float* We1 = (const float*)d_in[9];  const float* be1 = (const float*)d_in[10];
  const float* Ws1 = (const float*)d_in[11]; const float* bs1 = (const float*)d_in[12];
  const float* lng = (const float*)d_in[13]; const float* lnb = (const float*)d_in[14];
  const float* Wq2 = (const float*)d_in[15]; const float* bq2 = (const float*)d_in[16];
  const float* Wk2 = (const float*)d_in[17]; const float* bk2 = (const float*)d_in[18];
  const float* Wv2 = (const float*)d_in[19]; const float* bv2 = (const float*)d_in[20];
  const float* We2 = (const float*)d_in[21]; const float* be2 = (const float*)d_in[22];
  const float* Ws2 = (const float*)d_in[23]; const float* bs2 = (const float*)d_in[24];
  const int* src = eidx;
  const int* dst = eidx + N_EDGES;
  float* out = (float*)d_out;

  // ---- workspace carve-up
  char* p = (char*)d_ws;
  auto take = [&](size_t bytes) { char* r = p; p += alignup(bytes); return r; };
  __bf16* xb      = (__bf16*)take((size_t)N_NODES * 128 * 2);
  __bf16* eab     = (__bf16*)take((size_t)N_EDGES * ED_F * 2);
  __bf16* Wfrag   = (__bf16*)take((size_t)4 * 32 * 32 * 16 * 2);
  float*  biasP   = (float*)take(512 * 4);
  __bf16* Wefrag  = (__bf16*)take((size_t)8 * 32 * 16 * 2);
  float*  nodebuf = (float*)take((size_t)N_NODES * 512 * 4);
  float*  alpha1  = (float*)take((size_t)N_EDGES * 4 * 4);
  unsigned* amax1 = (unsigned*)take((size_t)N_NODES * 4 * 4);
  float*  denom1  = (float*)take((size_t)N_NODES * 4 * 4);
  float*  acc1    = (float*)take((size_t)N_NODES * 128 * 4);
  float*  hbuf    = (float*)take((size_t)N_NODES * 128 * 4);
  float*  q2 = (float*)take((size_t)N_NODES * 4);
  float*  k2 = (float*)take((size_t)N_NODES * 4);
  float*  v2 = (float*)take((size_t)N_NODES * 4);
  float*  s2 = (float*)take((size_t)N_NODES * 4);
  float*  e2buf  = (float*)take((size_t)N_EDGES * 4);
  float*  alpha2 = (float*)take((size_t)N_EDGES * 4);
  unsigned* amax2 = (unsigned*)take((size_t)N_NODES * 4);
  float*  denom2  = (float*)take((size_t)N_NODES * 4);
  float*  acc2    = (float*)take((size_t)N_NODES * 4);

  // ---- zero accumulators (enc(-inf) > 0 makes 0 a valid -inf for amax)
  hipMemsetAsync(amax1,  0, (size_t)N_NODES * 4 * 4, stream);
  hipMemsetAsync(denom1, 0, (size_t)N_NODES * 4 * 4, stream);
  hipMemsetAsync(acc1,   0, (size_t)N_NODES * 128 * 4, stream);
  hipMemsetAsync(amax2,  0, (size_t)N_NODES * 4, stream);
  hipMemsetAsync(denom2, 0, (size_t)N_NODES * 4, stream);
  hipMemsetAsync(acc2,   0, (size_t)N_NODES * 4, stream);

  // ---- prep
  cvt_bf16<<<8192, 256, 0, stream>>>(x,  xb,  N_NODES * 128);
  cvt_bf16<<<8192, 256, 0, stream>>>(ea, eab, N_EDGES * ED_F);
  pack_weights_l1<<<64, 256, 0, stream>>>(Wq1, Wk1, Wv1, Ws1, bq1, bk1, bv1, bs1, We1,
                                          Wfrag, biasP, Wefrag);
  // ---- layer 1
  node_gemm_l1<<<N_NODES / 16, 256, 0, stream>>>(xb, Wfrag, biasP, nodebuf);
  edge_attn_l1<<<2048, 128, 0, stream>>>(eab, Wefrag, be1, src, dst, nodebuf, alpha1, amax1);
  edge_agg_l1<<<2048, 128, 0, stream>>>(eab, Wefrag, be1, src, dst, nodebuf,
                                        alpha1, amax1, denom1, acc1);
  node_finish_l1<<<(N_NODES * 32) / 256, 256, 0, stream>>>(acc1, denom1, nodebuf, lng, lnb, hbuf);
  // ---- layer 2
  node_lin_l2<<<(N_NODES * 32) / 256, 256, 0, stream>>>(hbuf, Wq2, bq2, Wk2, bk2, Wv2, bv2,
                                                        Ws2, bs2, q2, k2, v2, s2);
  edge_l2_p1<<<N_EDGES / 256, 256, 0, stream>>>(ea, We2, be2, src, dst, q2, k2,
                                                e2buf, alpha2, amax2);
  edge_l2_p2<<<N_EDGES / 256, 256, 0, stream>>>(src, dst, v2, e2buf, alpha2, amax2,
                                                denom2, acc2);
  node_final<<<(N_NODES + 255) / 256, 256, 0, stream>>>(acc2, denom2, s2, out);
}
